// GraphAttentionLayer_88192858456650
// MI455X (gfx1250) — matrix operand order repaired and checked by reading, not compiled, not run on hardware
//
#include <hip/hip_runtime.h>

#define ALPHA   0.2f
#define NEG_INF -9.0e15f
#define HS 72          // H LDS row stride (bf16 elems) = 144B rows: 16B-aligned, conflict-free
#define PS 72          // P LDS row stride

typedef __attribute__((ext_vector_type(16))) __bf16       v16bf;
typedef __attribute__((ext_vector_type(8)))  float        v8f;
typedef __attribute__((ext_vector_type(8)))  unsigned int v8u;
typedef __attribute__((ext_vector_type(4)))  unsigned int u32x4;
typedef __attribute__((ext_vector_type(8)))  int          i32x8;
typedef __attribute__((ext_vector_type(4)))  int          i32x4;

__device__ __forceinline__ unsigned short f2bf(float x) {
    unsigned int u = __float_as_uint(x);
    u += 0x7FFFu + ((u >> 16) & 1u);          // round-to-nearest-even
    return (unsigned short)(u >> 16);
}
__device__ __forceinline__ float lrelu(float x) { return x > 0.f ? x : ALPHA * x; }

// TDM: DMA a 128(d) x 64(k) bf16 tile from hT (row stride N) into LDS with
// hardware row padding: 32 DWORDs (128B) data + 4 DWORDs (16B) pad = 144B rows.
__device__ __forceinline__ void tdm_load_tile(const unsigned short* gptr,
                                              unsigned ldsAddr, int N) {
    unsigned long long ga = (unsigned long long)gptr;
    u32x4 g0;
    g0[0] = 1u;                                        // count=1, user mode
    g0[1] = ldsAddr;                                   // lds_addr
    g0[2] = (unsigned)ga;                              // global_addr[31:0]
    g0[3] = (unsigned)(ga >> 32) | (2u << 30);         // global_addr[56:32] | type=2
    i32x8 g1;
    g1[0] = (int)((1u << 16)       // data_size = 2B
                | (1u << 20)       // pad_enable
                | (4u << 22)       // pad_interval = 32 DWORDs (128B)
                | (3u << 25));     // pad_amount   = 4 DWORDs (16B)
    g1[1] = (int)(((unsigned)N & 0xffffu) << 16);      // tensor_dim0[15:0]
    g1[2] = (int)(((unsigned)N >> 16) | (128u << 16)); // dim0[31:16] | tensor_dim1=128
    g1[3] = (int)(64u << 16);                          // tile_dim0 = 64
    g1[4] = (int)128u;                                 // tile_dim1 = 128 (tile_dim2=0)
    g1[5] = (int)N;                                    // tensor_dim0_stride[31:0]
    g1[6] = 0;
    g1[7] = 0;
    i32x4 gz4 = {0, 0, 0, 0};
    i32x8 gz8 = {0, 0, 0, 0, 0, 0, 0, 0};
    __builtin_amdgcn_tensor_load_to_lds(g0, g1, gz4, gz4, gz8, 0);
}

// ---------- Kernel 1: f1 = h.a1, f2 = h.a2 (one wave32 per row) -------------
__global__ void gat_proj(const float* __restrict__ h,
                         const float* __restrict__ a1,
                         const float* __restrict__ a2,
                         float* __restrict__ f1, float* __restrict__ f2,
                         int BN) {
    int wid = threadIdx.x >> 5, lane = threadIdx.x & 31;
    int gr = blockIdx.x * 4 + wid;
    if (gr >= BN) return;
    const float* hp = h + (size_t)gr * 128;
    float s1 = 0.f, s2 = 0.f;
#pragma unroll
    for (int q = 0; q < 4; ++q) {
        int d = lane + q * 32;
        float hv = hp[d];
        s1 += hv * a1[d];
        s2 += hv * a2[d];
    }
#pragma unroll
    for (int o = 16; o > 0; o >>= 1) {
        s1 += __shfl_xor(s1, o, 32);
        s2 += __shfl_xor(s2, o, 32);
    }
    if (lane == 0) { f1[gr] = s1; f2[gr] = s2; }
}

// ---------- Kernel 2: bf16 transpose h[b,i,d] -> hT[b,d,i] (TDM source) -----
__global__ void h_to_bf16T(const float* __restrict__ h,
                           unsigned short* __restrict__ hT, int N) {
    __shared__ unsigned short t[32][33];
    int b = blockIdx.z, i0 = blockIdx.x * 32, d0 = blockIdx.y * 32;
    int li = threadIdx.x, lj = threadIdx.y;   // 32 x 8
#pragma unroll
    for (int r = lj; r < 32; r += 8)
        t[r][li] = f2bf(h[((size_t)(b * N + i0 + r)) * 128 + d0 + li]);
    __syncthreads();
#pragma unroll
    for (int r = lj; r < 32; r += 8)
        hT[((size_t)(b * 128 + d0 + r)) * N + i0 + li] = t[li][r];
}

// ---------- Kernel 3: row softmax stats m,l (one wave32 per row) ------------
__global__ void gat_stats(const int* __restrict__ adj,
                          const float* __restrict__ f1,
                          const float* __restrict__ f2,
                          float* __restrict__ mrow, float* __restrict__ lrow,
                          int N) {
    int wid = threadIdx.x >> 5, lane = threadIdx.x & 31;
    int gr = blockIdx.x * 4 + wid;
    int b  = gr / N;
    const int*   ar  = adj + (size_t)gr * N;
    const float* f2b = f2 + (size_t)b * N;
    float f1i = f1[gr];
    float m = -3.0e38f, s = 0.f;
    for (int j = lane; j < N; j += 32) {
        float sc = (ar[j] > 0) ? lrelu(f1i + f2b[j]) : NEG_INF;
        float nm = fmaxf(m, sc);
        s = s * __expf(m - nm) + __expf(sc - nm);
        m = nm;
    }
#pragma unroll
    for (int o = 16; o > 0; o >>= 1) {
        float om = __shfl_xor(m, o, 32);
        float os = __shfl_xor(s, o, 32);
        float nm = fmaxf(m, om);
        s = s * __expf(m - nm) + os * __expf(om - nm);
        m = nm;
    }
    if (lane == 0) { mrow[gr] = m; lrow[gr] = s; }
}

// ---------- Kernel 4: fused  out = elu( (exp(score-m) @ h) / l ) ------------
// 4 waves/WG; wave = 16 rows x D=128. K-tile = 64 per stage.
// Double-buffered H tiles; TDM runs 2 iterations ahead of the WMMA consumer.
__global__ void __launch_bounds__(128, 1)
gat_attn(const unsigned short* __restrict__ hT,
         const int* __restrict__ adj,
         const float* __restrict__ f1w,
         const float* __restrict__ f2w,
         const float* __restrict__ mw,
         const float* __restrict__ lw,
         float* __restrict__ out, int N) {
    __shared__ unsigned short Hlds[2][128 * HS];    // double-buffered TDM dest
    __shared__ unsigned short Plds[4 * 16 * PS];    // per-wave P tile [r][k]

    const int tid = threadIdx.x, wid = tid >> 5, lane = tid & 31;
    const int tilesPerB = N >> 6;                   // 64 rows per WG
    const int b = blockIdx.x / tilesPerB, tile = blockIdx.x % tilesPerB;
    const int growbase = b * N + tile * 64 + wid * 16;

    const int Mlane = lane & 15, half = lane >> 4;
    unsigned short* Pw = Plds + wid * (16 * PS);

    // P staging: lane owns row Mlane, columns [half*32, half*32+32)
    const int cbase = half * 32;
    const float f1v = f1w[growbase + Mlane];        // 1 scalar per lane (was 16)
    const float mv  = mw[growbase + Mlane];
    const int*   aprow = adj + (size_t)(growbase + Mlane) * N + cbase;
    const float* fprow = f2w + (size_t)b * N + cbase;
    unsigned short* Pwr = Pw + Mlane * PS + cbase;

    v8f acc[8] = {};

    const unsigned ldsB0 = (unsigned)(unsigned long long)&Hlds[0][0];
    const unsigned ldsB1 = (unsigned)(unsigned long long)&Hlds[1][0];
    const unsigned short* tileBase = hT + (size_t)b * 128 * N;
    const int iters = N >> 6;

    if (wid == 0) {                       // prime the 2-deep TDM pipeline
        tdm_load_tile(tileBase, ldsB0, N);
        if (iters > 1) tdm_load_tile(tileBase + 64, ldsB1, N);
    }

    for (int i = 0; i < iters; ++i) {
        const int k0 = i << 6;
        const unsigned short* Hb = &Hlds[i & 1][0];

        // ---- P tile (16 x 64): overlaps in-flight TDM; int4 adj loads ----
#pragma unroll
        for (int c = 0; c < 32; c += 4) {
            int4   a4 = *(const int4*)(aprow + k0 + c);
            float4 f4 = *(const float4*)(fprow + k0 + c);
            float p0 = __expf(((a4.x > 0) ? lrelu(f1v + f4.x) : NEG_INF) - mv);
            float p1 = __expf(((a4.y > 0) ? lrelu(f1v + f4.y) : NEG_INF) - mv);
            float p2 = __expf(((a4.z > 0) ? lrelu(f1v + f4.z) : NEG_INF) - mv);
            float p3 = __expf(((a4.w > 0) ? lrelu(f1v + f4.w) : NEG_INF) - mv);
            *(unsigned*)&Pwr[c]     = (unsigned)f2bf(p0) | ((unsigned)f2bf(p1) << 16);
            *(unsigned*)&Pwr[c + 2] = (unsigned)f2bf(p2) | ((unsigned)f2bf(p3) << 16);
        }
        if (wid == 0) {                    // tile i landed (in-order completion)
            if (i + 1 < iters) __builtin_amdgcn_s_wait_tensorcnt(1);
            else               __builtin_amdgcn_s_wait_tensorcnt(0);
        }
        __syncthreads();

#pragma unroll
        for (int kh = 0; kh < 2; ++kh) {
            const int kb = kh * 32;
            // A fragment (16x32): lanes 0-15 M=lane K {0..7,16..23}; lanes 16-31 K+8
            v8u au;
#pragma unroll
            for (int v = 0; v < 8; ++v) {
                int K = kb + ((v < 4) ? 2 * v : 2 * v + 8) + 8 * half;
                au[v] = *(const unsigned int*)&Pw[Mlane * PS + K];
            }
            v16bf afrag = __builtin_bit_cast(v16bf, au);
            v16bf bfrag[8];
#pragma unroll
            for (int dt = 0; dt < 8; ++dt) {
                v8u bu;
                int col = dt * 16 + Mlane;
#pragma unroll
                for (int v = 0; v < 8; ++v) {
                    int K = kb + ((v < 4) ? 2 * v : 2 * v + 8) + 8 * half;
                    bu[v] = *(const unsigned int*)&Hb[col * HS + K];
                }
                bfrag[dt] = __builtin_bit_cast(v16bf, bu);
            }
#pragma unroll
            for (int dt = 0; dt < 8; ++dt)
                acc[dt] = __builtin_amdgcn_wmma_f32_16x16x32_bf16(
                    false, afrag, false, bfrag[dt], (short)0, acc[dt], false, false);
        }
        __syncthreads();   // all waves done reading Hlds[i&1] -> safe to refill

        if (wid == 0 && i + 2 < iters)
            tdm_load_tile(tileBase + (size_t)(k0 + 128),
                          (i & 1) ? ldsB1 : ldsB0, N);
    }

    // ---- epilogue: lanes 0-15 M=g, lanes 16-31 M=8+g; N col = Mlane ----
#pragma unroll
    for (int g = 0; g < 8; ++g) {
        int row = growbase + g + 8 * half;
        float linv = 1.0f / lw[row];
#pragma unroll
        for (int dt = 0; dt < 8; ++dt) {
            float v = acc[dt][g] * linv;
            v = (v > 0.f) ? v : (__expf(v) - 1.0f);   // ELU
            out[(size_t)row * 128 + dt * 16 + Mlane] = v;
        }
    }
}

extern "C" void kernel_launch(void* const* d_in, const int* in_sizes, int n_in,
                              void* d_out, int out_size, void* d_ws, size_t ws_size,
                              hipStream_t stream) {
    const float* hG   = (const float*)d_in[0];   // [B,N,D] f32
    const int*   adjG = (const int*)d_in[1];     // [B,N,N] i32
    const float* a1   = (const float*)d_in[2];   // [D,1]
    const float* a2   = (const float*)d_in[3];   // [D,1]
    float* outG = (float*)d_out;

    const int D  = in_sizes[2];                  // 128
    const int BN = in_sizes[0] / D;              // B*N
    const int N  = in_sizes[1] / BN;             // N
    const int B  = BN / N;

    float* f1 = (float*)d_ws;                    // BN f32 each
    float* f2 = f1 + BN;
    float* mw = f2 + BN;
    float* lw = mw + BN;
    unsigned short* hT = (unsigned short*)(lw + BN);   // BN*D bf16, [b][d][i]

    gat_proj  <<<(BN + 3) / 4, 128, 0, stream>>>(hG, a1, a2, f1, f2, BN);
    h_to_bf16T<<<dim3(N / 32, D / 32, B), dim3(32, 8), 0, stream>>>(hG, hT, N);
    gat_stats <<<(BN + 3) / 4, 128, 0, stream>>>(adjG, f1, f2, mw, lw, N);
    gat_attn  <<<BN / 64, 128, 0, stream>>>(hT, adjG, f1, f2, mw, lw, outG, N);
}
